// PhraseClassifier_13993003450895
// MI455X (gfx1250) — matrix operand order repaired and checked by reading, not compiled, not run on hardware
//
#include <hip/hip_runtime.h>
#include <hip/hip_bf16.h>

// ---------------------------------------------------------------------------
// Problem constants (from reference): B=4, S=256, H=768, N=128, HID=1024, L=16
// ---------------------------------------------------------------------------
#define Bb   4
#define Ss   256
#define Hh   768
#define Nn   128
#define HIDh 1024
#define Ll   16

typedef __attribute__((ext_vector_type(16))) __bf16          v16bf;
typedef __attribute__((ext_vector_type(8)))  float           v8f;
typedef __attribute__((ext_vector_type(4)))  unsigned int    v4u;
typedef __attribute__((ext_vector_type(8)))  unsigned int    v8u;

__device__ __forceinline__ float bf2f(unsigned short u) {
    union { unsigned int i; float f; } c; c.i = ((unsigned int)u) << 16; return c.f;
}
__device__ __forceinline__ unsigned short f2bf(float f) {
    union { float f; unsigned int i; } c; c.f = f;
    unsigned int u = c.i;
    u += 0x7fffu + ((u >> 16) & 1u);      // round-to-nearest-even
    return (unsigned short)(u >> 16);
}

// Packed bf16 multiply: 2 elements/instruction, RNE, no f32 round-trip.
// (no clang builtin exposes V_PK_MUL_BF16 -> inline asm per CDNA5 VOP3P)
__device__ __forceinline__ unsigned int pkmul_bf16(unsigned int a, unsigned int b) {
    unsigned int d;
    asm("v_pk_mul_bf16 %0, %1, %2" : "=v"(d) : "v"(a), "v"(b));
    return d;
}

// Fast tanh via hardware transcendentals (v_exp_f32 + v_rcp_f32).
__device__ __forceinline__ float tanh_fast(float x) {
    x = fminf(fmaxf(x, -15.f), 15.f);
    float t = __builtin_amdgcn_exp2f(x * 2.885390081777927f);  // e^(2x)
    float r = __builtin_amdgcn_rcpf(t + 1.f);
    return (t - 1.f) * r;
}

// ---------------------------------------------------------------------------
// K1: gather con[b,n,:] = all_hidden[b, starts[b,n], :]  -> bf16
// ---------------------------------------------------------------------------
__global__ void k_gather(const float* __restrict__ ah,
                         const int*   __restrict__ starts,
                         unsigned short* __restrict__ conb)
{
    int t = blockIdx.x * 256 + threadIdx.x;        // over B*N*H
    if (t >= Bb * Nn * Hh) return;
    int h  = t % Hh;
    int bn = t / Hh;
    int n  = bn % Nn;
    int b  = bn / Nn;
    int s  = starts[b * Nn + n];
    conb[t] = f2bf(ah[((size_t)b * Ss + s) * Hh + h]);
}

// ---------------------------------------------------------------------------
// K2: repack W1d (rows 2304..3071 of W1) and W2 into bf16 WMMA-B layouts:
//   wdt[ks][h][kk]  ks=0..23 (K-slab of 32), h=0..1023, kk=0..31
//   w2t[kt][l][kk]  kt=0..31, l=0..15, kk=0..31
// ---------------------------------------------------------------------------
__global__ void k_repack(const float* __restrict__ W1,
                         const float* __restrict__ W2,
                         unsigned short* __restrict__ wdt,
                         unsigned short* __restrict__ w2t)
{
    int t = blockIdx.x * 256 + threadIdx.x;
    const int nd = Hh * HIDh;                       // 768*1024
    if (t < nd) {
        int h = t % HIDh, k = t / HIDh;             // k = 0..767
        int ks = k >> 5, kk = k & 31;
        wdt[((size_t)(ks * HIDh + h)) * 32 + kk] =
            f2bf(W1[(size_t)(3 * Hh + k) * HIDh + h]);
    } else {
        int e = t - nd;
        if (e < HIDh * Ll) {
            int l = e % Ll, k = e / Ll;             // k = 0..1023
            int kt = k >> 5, kk = k & 31;
            w2t[((size_t)(kt * Ll + l)) * 32 + kk] = f2bf(W2[(size_t)k * Ll + l]);
        }
    }
}

// ---------------------------------------------------------------------------
// K3: P[bn,h] = con[bn]·(W1a+W1c)[:,h],  Q[bn,h] = con[bn]·(W1b-W1c)[:,h]
// ---------------------------------------------------------------------------
__global__ __launch_bounds__(256) void k_pq(const unsigned short* __restrict__ conb,
                                            const float* __restrict__ W1,
                                            float* __restrict__ P,
                                            float* __restrict__ Q)
{
    __shared__ float arow[Hh];
    const int bn  = blockIdx.x;                     // 0..511
    const int tid = threadIdx.x;
    const unsigned short* row = conb + (size_t)bn * Hh;
    for (int k = tid; k < Hh; k += 256) arow[k] = bf2f(row[k]);
    __syncthreads();
    for (int hh = 0; hh < 4; ++hh) {
        int h = hh * 256 + tid;
        float p = 0.f, q = 0.f;
        #pragma unroll 4
        for (int k = 0; k < Hh; ++k) {
            float a  = arow[k];
            float wa = W1[(size_t)k * HIDh + h];
            float wb = W1[(size_t)(Hh + k) * HIDh + h];
            float wc = W1[(size_t)(2 * Hh + k) * HIDh + h];
            p += a * (wa + wc);
            q += a * (wb - wc);
        }
        P[(size_t)bn * HIDh + h] = p;
        Q[(size_t)bn * HIDh + h] = q;
    }
}

// ---------------------------------------------------------------------------
// K4: main. One WG per (b,i). 8 waves; wave w owns h-strip [w*128, w*128+128).
// ---------------------------------------------------------------------------
__global__ __launch_bounds__(256) void k_main(const unsigned short* __restrict__ conb,
                                              const unsigned short* __restrict__ wdt,
                                              const unsigned short* __restrict__ w2t,
                                              const float* __restrict__ P,
                                              const float* __restrict__ Q,
                                              const float* __restrict__ b1,
                                              const float* __restrict__ b2,
                                              float* __restrict__ out)
{
    __shared__ unsigned short hidl[8][16 * 128];    // per-wave bf16 hid strip (32 KB)
    __shared__ float pout[8][16][16];               // per-wave partial outputs (8 KB)

    const int bi   = blockIdx.x;                    // 0..511 = (b*128 + i)
    const int b    = bi >> 7;
    const int i    = bi & 127;
    const int tid  = threadIdx.x;
    const int w    = tid >> 5;                      // wave id (wave32)
    const int lane = tid & 31;
    const int lh   = lane & 15;                     // row/col within fragment
    const int kh   = lane >> 4;                     // K-half select
    const int mhi  = kh * 8;                        // C/D row offset for hi lanes

    const unsigned short* conB = conb + (size_t)b * Nn * Hh;
    const unsigned short* rowI = conB + (size_t)i * Hh;

    // jt-invariant: P[b,i,h] + b1[h] for this lane's 8 h-columns
    float pb[8];
    #pragma unroll
    for (int nt = 0; nt < 8; ++nt) {
        const int h = w * 128 + nt * 16 + lh;
        pb[nt] = P[((size_t)bi) * HIDh + h] + b1[h];
    }

    for (int jt = 0; jt < 8; ++jt) {
        v8f acc[8];
        #pragma unroll
        for (int nt = 0; nt < 8; ++nt)
            #pragma unroll
            for (int e = 0; e < 8; ++e) acc[nt][e] = 0.f;

        const int jrow = jt * 16 + lh;              // A-fragment row for this lane
        const unsigned short* rowJ = conB + (size_t)jrow * Hh;

        for (int ks = 0; ks < 24; ++ks) {
            const int k0 = ks * 32;
            // prefetch next K-slab of the B stream (global_prefetch_b8)
            if (ks + 1 < 24)
                __builtin_prefetch(wdt + ((size_t)((ks + 1) * HIDh + w * 128)) * 32, 0, 0);
            // ---- A fragment: A[j,k] = bf16(con_j[k] * con_i[k]) via pk_mul --
            v8u a0 = *(const v8u*)(const void*)(rowJ + k0 + kh * 16);
            v8u s0 = *(const v8u*)(const void*)(rowI + k0 + kh * 16);
            v8u av;
            #pragma unroll
            for (int e = 0; e < 8; ++e) {
                av[e]     = pkmul_bf16(a0[e], s0[e]);
            }
            v16bf A = __builtin_bit_cast(v16bf, av);
            // ---- 8 N-tiles of 16 columns in this wave's h-strip -------------
            #pragma unroll
            for (int nt = 0; nt < 8; ++nt) {
                const int h = w * 128 + nt * 16 + lh;
                const unsigned short* bp =
                    wdt + ((size_t)(ks * HIDh + h)) * 32 + kh * 16;
                v16bf Bf = __builtin_bit_cast(v16bf, *(const v8u*)(const void*)bp);
                acc[nt] = __builtin_amdgcn_wmma_f32_16x16x32_bf16(
                    false, A, false, Bf, (short)0, acc[nt], false, false);
            }
        }

        // ---- epilogue 1: tanh(S + P + Q + b1), spill bf16 strip to LDS ------
        #pragma unroll
        for (int nt = 0; nt < 8; ++nt) {
            const int h = w * 128 + nt * 16 + lh;
            #pragma unroll
            for (int r = 0; r < 8; ++r) {
                int jr  = jt * 16 + r + mhi;
                float q = Q[((size_t)(b * Nn + jr)) * HIDh + h];
                float v = tanh_fast(acc[nt][r] + pb[nt] + q);
                hidl[w][(r + mhi) * 128 + nt * 16 + lh] = f2bf(v);
            }
        }
        // same-wave LDS store->load is in-order; no barrier needed here

        // ---- epilogue 2: partial out = hid_strip @ W2_strip via WMMA --------
        v8f acc2;
        #pragma unroll
        for (int e = 0; e < 8; ++e) acc2[e] = 0.f;
        #pragma unroll
        for (int kt = 0; kt < 4; ++kt) {
            const unsigned short* hb = &hidl[w][lh * 128 + kt * 32 + kh * 16];
            v8u av2 = *(const v8u*)(const void*)hb;
            v16bf A2 = __builtin_bit_cast(v16bf, av2);
            const unsigned short* wp =
                w2t + ((size_t)((w * 4 + kt) * Ll + lh)) * 32 + kh * 16;
            v16bf B2 = __builtin_bit_cast(v16bf, *(const v8u*)(const void*)wp);
            acc2 = __builtin_amdgcn_wmma_f32_16x16x32_bf16(
                false, A2, false, B2, (short)0, acc2, false, false);
        }
        #pragma unroll
        for (int r = 0; r < 8; ++r) pout[w][r + mhi][lh] = acc2[r];
        __syncthreads();

        // ---- cross-wave reduce + write out (256 threads == 16x16 outputs) ---
        {
            int r = tid >> 4, l = tid & 15;
            float s = 0.f;
            #pragma unroll
            for (int ww = 0; ww < 8; ++ww) s += pout[ww][r][l];
            s += b2[l];
            size_t o = (((size_t)bi) * Nn + (jt * 16 + r)) * Ll + l;
            out[o] = s;
        }
        __syncthreads();                            // protect pout for next j-tile
    }
}

// ---------------------------------------------------------------------------
// Launch
// ---------------------------------------------------------------------------
static size_t align256(size_t x) { return (x + 255) & ~(size_t)255; }

extern "C" void kernel_launch(void* const* d_in, const int* in_sizes, int n_in,
                              void* d_out, int out_size, void* d_ws, size_t ws_size,
                              hipStream_t stream) {
    const float* ah     = (const float*)d_in[0];   // [B,S,H]
    const int*   starts = (const int*)  d_in[1];   // [B,N]
    const float* W1     = (const float*)d_in[2];   // [3072,1024]
    const float* b1     = (const float*)d_in[3];   // [1024]
    const float* W2     = (const float*)d_in[4];   // [1024,16]
    const float* b2     = (const float*)d_in[5];   // [16]
    float*       out    = (float*)d_out;           // [B,N,N,L]

    char* ws = (char*)d_ws;
    unsigned short* conb = (unsigned short*)ws; ws += align256((size_t)Bb * Nn * Hh * 2);
    unsigned short* wdt  = (unsigned short*)ws; ws += align256((size_t)24 * HIDh * 32 * 2);
    unsigned short* w2t  = (unsigned short*)ws; ws += align256((size_t)32 * Ll * 32 * 2);
    float*          Pt   = (float*)ws;          ws += align256((size_t)Bb * Nn * HIDh * 4);
    float*          Qt   = (float*)ws;          ws += align256((size_t)Bb * Nn * HIDh * 4);

    {   // K1: gather + bf16 convert
        int total = Bb * Nn * Hh;
        k_gather<<<(total + 255) / 256, 256, 0, stream>>>(ah, starts, conb);
    }
    {   // K2: weight repack into WMMA B-fragment layout
        int total = Hh * HIDh + HIDh * Ll;
        k_repack<<<(total + 255) / 256, 256, 0, stream>>>(W1, W2, wdt, w2t);
    }
    {   // K3: P/Q rank-N tables
        k_pq<<<Bb * Nn, 256, 0, stream>>>(conb, W1, Pt, Qt);
    }
    {   // K4: main WMMA kernel, one WG per (b,i)
        k_main<<<Bb * Nn, 256, 0, stream>>>(conb, wdt, w2t, Pt, Qt, b1, b2, out);
    }
}
